// MyGraphUNet_3332894621893
// MI455X (gfx1250) — matrix operand order repaired and checked by reading, not compiled, not run on hardware
//
#include <hip/hip_runtime.h>

// ---------------------------------------------------------------------------
// MyGraphUNet on gfx1250 (MI455X): GroupNorm+GCNx2+time-embed+residual+GAT.
// GEMMs use v_wmma_f32_16x16x32_f16 (wave32 WMMA) with pre-packed B fragments
// (one contiguous 32B load per lane) and 4-way M-blocking per wave.
// Edge aggregation: float4 gathers + global_atomic_add_f32; segment-max via
// monotone-uint atomicMax. All scratch lives in d_ws (~105 MB with reuse).
// ---------------------------------------------------------------------------

typedef __attribute__((ext_vector_type(16))) _Float16 v16h;
typedef __attribute__((ext_vector_type(8)))  _Float16 v8h;
typedef __attribute__((ext_vector_type(8)))  float    v8f;

#define CIN   64
#define COUT  128
#define TCH   512
#define NHEAD 4
#define DHEAD 32

// ------------------------------ small utils -------------------------------

// tt[c] = te_b[c] + sum_k leaky(t[k]) * te_w[k,c]
__global__ void k_time_embed(const float* __restrict__ t, const float* __restrict__ W,
                             const float* __restrict__ b, float* __restrict__ tt) {
  int c = threadIdx.x;                      // 128 threads
  float acc = b[c];
  for (int k = 0; k < TCH; ++k) {
    float v = t[k];
    v = v > 0.f ? v : 0.01f * v;
    acc += v * W[k * COUT + c];
  }
  tt[c] = acc;
}

__global__ void k_deg_init(float* __restrict__ deg, unsigned* __restrict__ fillb, int N) {
  int i = blockIdx.x * blockDim.x + threadIdx.x;
  if (i < N) deg[i] = 1.0f;                 // self-loop weight 1
  if (i == 0) *fillb = 0u;
}

__global__ void k_deg_edge(const int* __restrict__ col, const float* __restrict__ ew,
                           float* __restrict__ deg, unsigned* __restrict__ fillb, int E) {
  int e = blockIdx.x * blockDim.x + threadIdx.x;
  if (e >= E) return;
  float w = ew[e];
  atomicAdd(&deg[col[e]], w);
  atomicMax(fillb, __float_as_uint(w));     // ew > 0 -> raw bits monotone
}

__global__ void k_dinv(float* __restrict__ deg, int N) {
  int i = blockIdx.x * blockDim.x + threadIdx.x;
  if (i >= N) return;
  float d = deg[i];
  deg[i] = d > 0.f ? rsqrtf(d) : 0.f;
}

// ------------------------------ group norms -------------------------------

// GN over 64 channels, 8 groups (8 ch/group), + leaky(0.01); also emits raw x f16.
__global__ void k_gn64(const float* __restrict__ x, const float* __restrict__ w,
                       const float* __restrict__ b, _Float16* __restrict__ xn16,
                       _Float16* __restrict__ x16, int N) {
  int t = blockIdx.x * 256 + threadIdx.x;
  int node = t >> 6, ch = t & 63;
  if (node >= N) return;
  float v = x[node * CIN + ch];
  if (x16) x16[node * CIN + ch] = (_Float16)v;
  float s = v, s2 = v * v;
#pragma unroll
  for (int m = 1; m < 8; m <<= 1) {         // groups of 8 lanes
    s  += __shfl_xor(s,  m, 32);
    s2 += __shfl_xor(s2, m, 32);
  }
  float mu  = s * 0.125f;
  float var = s2 * 0.125f - mu * mu;
  float y = (v - mu) * rsqrtf(var + 1e-5f) * w[ch] + b[ch];
  y = y > 0.f ? y : 0.01f * y;
  xn16[node * CIN + ch] = (_Float16)y;
}

// GN over 128 channels, 8 groups (16 ch/group), + leaky(0.01) -> f16
__global__ void k_gn128(const float* __restrict__ h, const float* __restrict__ w,
                        const float* __restrict__ b, _Float16* __restrict__ out16, int N) {
  int t = blockIdx.x * 256 + threadIdx.x;
  int node = t >> 7, ch = t & 127;
  if (node >= N) return;
  float v = h[node * COUT + ch];
  float s = v, s2 = v * v;
#pragma unroll
  for (int m = 1; m < 16; m <<= 1) {        // groups of 16 lanes
    s  += __shfl_xor(s,  m, 32);
    s2 += __shfl_xor(s2, m, 32);
  }
  float mu  = s * (1.f / 16.f);
  float var = s2 * (1.f / 16.f) - mu * mu;
  float y = (v - mu) * rsqrtf(var + 1e-5f) * w[ch] + b[ch];
  y = y > 0.f ? y : 0.01f * y;
  out16[node * COUT + ch] = (_Float16)y;
}

// ------------------------------ WMMA GEMM ---------------------------------
// B fragment layout (CDNA5 16-bit B 32x16): lane L holds col n = L&15;
// element e<8 maps to K-offset (L>>4)*8+e, e>=8 to 16+(L>>4)*8+(e-8) (matches A);
// we pre-pack B so each lane's 16 halfs are
// contiguous: Bp[((nt*KT + kt)*32 + lane)*16 + e].
__global__ void k_pack_b(const float* __restrict__ B, _Float16* __restrict__ Bp,
                         int K, int Nc) {
  int t = blockIdx.x * blockDim.x + threadIdx.x;
  if (t >= K * Nc) return;
  int e    = t & 15;
  int lane = (t >> 4) & 31;
  int rest = t >> 9;                 // nt*KT + kt
  int KT   = K >> 5;
  int kt   = rest % KT;
  int nt   = rest / KT;
  int n = nt * 16 + (lane & 15);
  int k = kt * 32 + (lane >> 4) * 8 + (e & 7) + (e >> 3) * 16;
  Bp[t] = (_Float16)B[k * Nc + n];
}

// A fragment (CDNA5 16-bit A 16x32): lane L holds row m = L&15; the lane's 16
// halfs live at row offsets [koff .. koff+7] and [koff+16 .. koff+23] where
// koff = kt*32 + (L>>4)*8.  Two aligned 16B loads.
__device__ __forceinline__ v16h load_a16(const _Float16* __restrict__ A, int lda,
                                         int m, int koff) {
  const _Float16* rp = A + m * lda + koff;
  v8h lo = *(const v8h*)(rp);
  v8h hi = *(const v8h*)(rp + 16);
  return __builtin_shufflevector(lo, hi, 0, 1, 2, 3, 4, 5, 6, 7,
                                         8, 9, 10, 11, 12, 13, 14, 15);
}

// C[MxNc] = A[MxK] @ Bp; f16 in, f32 out. blockDim=128 (4 waves, each one
// n-tile x four m-tiles); grid = (ceil(M/64), Nc/64). K compile-time.
template <int K>
__global__ void k_gemm_wmma(const _Float16* __restrict__ A, const _Float16* __restrict__ Bp,
                            float* __restrict__ C, int M, int Nc) {
  constexpr int KT = K / 32;
  int lane = threadIdx.x & 31;
  int wave = threadIdx.x >> 5;
  int m0 = blockIdx.x * 64;
  int nt = blockIdx.y * 4 + wave;
  if (nt * 16 >= Nc) return;
  int kh = lane >> 4;

  int mr[4];
#pragma unroll
  for (int i = 0; i < 4; ++i) {
    int m = m0 + i * 16 + (lane & 15);
    mr[i] = m < M ? m : M - 1;               // clamp (stores are guarded)
  }

  v8f acc[4];
#pragma unroll
  for (int i = 0; i < 4; ++i) acc[i] = (v8f){0.f, 0.f, 0.f, 0.f, 0.f, 0.f, 0.f, 0.f};

  const _Float16* bbase = Bp + ((size_t)nt * KT * 32 + lane) * 16;
#pragma unroll
  for (int kt = 0; kt < KT; ++kt) {
    v16h b = *(const v16h*)(bbase + (size_t)kt * 32 * 16);
    int koff = kt * 32 + kh * 8;
#pragma unroll
    for (int i = 0; i < 4; ++i) {
      v16h a = load_a16(A, K, mr[i], koff);
      acc[i] = __builtin_amdgcn_wmma_f32_16x16x32_f16(false, a, false, b,
                                                      (short)0, acc[i], false, false);
    }
  }

  // C/D layout: element j -> row mtile + 8*(lane>>4) + j, col n0 + (lane&15)
  int n = nt * 16 + (lane & 15);
#pragma unroll
  for (int i = 0; i < 4; ++i) {
    int mb = m0 + i * 16 + 8 * kh;
#pragma unroll
    for (int j = 0; j < 8; ++j) {
      int m = mb + j;
      if (m < M) C[(size_t)m * Nc + n] = acc[i][j];
    }
  }
}

// ------------------------------ GCN aggregation ---------------------------

// acc[i,c] = dinv[i]^2 * xw[i,c] + bias[c] (+ addvec[c]) (+ addnode[i,c])
__global__ void k_gcn_init(float* __restrict__ acc, const float* __restrict__ xw,
                           const float* __restrict__ dinv, const float* __restrict__ bias,
                           const float* __restrict__ addvec, const float* __restrict__ addnode,
                           int N) {
  int t = blockIdx.x * blockDim.x + threadIdx.x;
  if (t >= N * COUT) return;
  int i = t >> 7, c = t & 127;
  float di = dinv[i];
  float v = di * di * xw[t] + bias[c];
  if (addvec)  v += addvec[c];
  if (addnode) v += addnode[t];
  acc[t] = v;
}

// per (edge, 4-channel chunk): acc[col] += dinv[r]*ew*dinv[c] * xw[row]
__global__ void k_gcn_edge(const int* __restrict__ row, const int* __restrict__ col,
                           const float* __restrict__ ew, const float* __restrict__ dinv,
                           const float* __restrict__ xw, float* __restrict__ acc, int E) {
  long long t = (long long)blockIdx.x * blockDim.x + threadIdx.x;
  if (t >= (long long)E * 32) return;
  int e = (int)(t >> 5), q = (int)(t & 31);
  int r = row[e], c = col[e];
  float coef = dinv[r] * ew[e] * dinv[c];
  const float4 v = *(const float4*)(xw + (size_t)r * COUT + 4 * q);
  float* dst = acc + (size_t)c * COUT + 4 * q;
  atomicAdd(dst + 0, coef * v.x);
  atomicAdd(dst + 1, coef * v.y);
  atomicAdd(dst + 2, coef * v.z);
  atomicAdd(dst + 3, coef * v.w);
}

// ------------------------------ GAT attention -----------------------------

__device__ __forceinline__ float lrelu02(float x) { return x > 0.f ? x : 0.2f * x; }
__device__ __forceinline__ unsigned encf(float f) {
  unsigned u = __float_as_uint(f);
  return (u & 0x80000000u) ? ~u : (u | 0x80000000u);
}
__device__ __forceinline__ float decf(unsigned e) {
  return (e & 0x80000000u) ? __uint_as_float(e ^ 0x80000000u) : __uint_as_float(~e);
}

// al_s[i,h], al_d[i,h] = sum_d hA[i,h*32+d] * a_{src,dst}[h,d]
__global__ void k_attn_coef(const float* __restrict__ hA, const float* __restrict__ asrc,
                            const float* __restrict__ adst, float* __restrict__ als,
                            float* __restrict__ ald, int N) {
  int t = blockIdx.x * blockDim.x + threadIdx.x;
  if (t >= N * NHEAD) return;
  int i = t >> 2, h = t & 3;
  const float* hp = hA + (size_t)i * COUT + h * DHEAD;
  const float* sp = asrc + h * DHEAD;
  const float* dp = adst + h * DHEAD;
  float as = 0.f, ad = 0.f;
#pragma unroll 4
  for (int d = 0; d < DHEAD; ++d) { float v = hp[d]; as += v * sp[d]; ad += v * dp[d]; }
  als[t] = as; ald[t] = ad;
}

__global__ void k_attn_minit(const float* __restrict__ als, const float* __restrict__ ald,
                             const float* __restrict__ aedge, const unsigned* __restrict__ fillb,
                             unsigned* __restrict__ mx, int N) {
  int t = blockIdx.x * blockDim.x + threadIdx.x;
  if (t >= N * NHEAD) return;
  int h = t & 3;
  float fill = __uint_as_float(*fillb);
  mx[t] = encf(lrelu02(als[t] + ald[t] + aedge[h] * fill));   // self-loop logit
}

__global__ void k_attn_medge(const int* __restrict__ row, const int* __restrict__ col,
                             const float* __restrict__ ew, const float* __restrict__ als,
                             const float* __restrict__ ald, const float* __restrict__ aedge,
                             unsigned* __restrict__ mx, int E) {
  int t = blockIdx.x * blockDim.x + threadIdx.x;
  if (t >= E * NHEAD) return;
  int e = t >> 2, h = t & 3;
  int r = row[e], c = col[e];
  float l = lrelu02(als[r * NHEAD + h] + ald[c * NHEAD + h] + aedge[h] * ew[e]);
  atomicMax(&mx[c * NHEAD + h], encf(l));
}

__global__ void k_attn_sinit(const float* __restrict__ als, const float* __restrict__ ald,
                             const float* __restrict__ aedge, const unsigned* __restrict__ fillb,
                             const unsigned* __restrict__ mx, float* __restrict__ s, int N) {
  int t = blockIdx.x * blockDim.x + threadIdx.x;
  if (t >= N * NHEAD) return;
  int h = t & 3;
  float fill = __uint_as_float(*fillb);
  float l = lrelu02(als[t] + ald[t] + aedge[h] * fill);
  s[t] = expf(l - decf(mx[t]));             // self-loop contribution
}

__global__ void k_attn_sedge(const int* __restrict__ row, const int* __restrict__ col,
                             const float* __restrict__ ew, const float* __restrict__ als,
                             const float* __restrict__ ald, const float* __restrict__ aedge,
                             const unsigned* __restrict__ mx, float* __restrict__ s, int E) {
  int t = blockIdx.x * blockDim.x + threadIdx.x;
  if (t >= E * NHEAD) return;
  int e = t >> 2, h = t & 3;
  int r = row[e], c = col[e];
  float l = lrelu02(als[r * NHEAD + h] + ald[c * NHEAD + h] + aedge[h] * ew[e]);
  atomicAdd(&s[c * NHEAD + h], expf(l - decf(mx[c * NHEAD + h])));
}

// out[i,c] = ba[c] + alpha_self * hA[i,c]
__global__ void k_attn_outinit(const float* __restrict__ hA, const float* __restrict__ als,
                               const float* __restrict__ ald, const float* __restrict__ aedge,
                               const unsigned* __restrict__ fillb, const unsigned* __restrict__ mx,
                               const float* __restrict__ s, const float* __restrict__ ba,
                               float* __restrict__ out, int N) {
  int t = blockIdx.x * blockDim.x + threadIdx.x;
  if (t >= N * COUT) return;
  int i = t >> 7, c = t & 127, h = c >> 5;
  float fill = __uint_as_float(*fillb);
  float l = lrelu02(als[i * NHEAD + h] + ald[i * NHEAD + h] + aedge[h] * fill);
  float alpha = expf(l - decf(mx[i * NHEAD + h])) / (s[i * NHEAD + h] + 1e-16f);
  out[t] = ba[c] + alpha * hA[t];
}

// per (edge, 4-channel chunk): out[col] += alpha[e,h] * hA[row]
__global__ void k_attn_edge(const int* __restrict__ row, const int* __restrict__ col,
                            const float* __restrict__ ew, const float* __restrict__ als,
                            const float* __restrict__ ald, const float* __restrict__ aedge,
                            const unsigned* __restrict__ mx, const float* __restrict__ s,
                            const float* __restrict__ hA, float* __restrict__ out, int E) {
  long long t = (long long)blockIdx.x * blockDim.x + threadIdx.x;
  if (t >= (long long)E * 32) return;
  int e = (int)(t >> 5), q = (int)(t & 31), h = q >> 3;
  int r = row[e], c = col[e];
  float l = lrelu02(als[r * NHEAD + h] + ald[c * NHEAD + h] + aedge[h] * ew[e]);
  float alpha = expf(l - decf(mx[c * NHEAD + h])) / (s[c * NHEAD + h] + 1e-16f);
  const float4 v = *(const float4*)(hA + (size_t)r * COUT + 4 * q);
  float* dst = out + (size_t)c * COUT + 4 * q;
  atomicAdd(dst + 0, alpha * v.x);
  atomicAdd(dst + 1, alpha * v.y);
  atomicAdd(dst + 2, alpha * v.z);
  atomicAdd(dst + 3, alpha * v.w);
}

// ------------------------------ launcher ----------------------------------

extern "C" void kernel_launch(void* const* d_in, const int* in_sizes, int n_in,
                              void* d_out, int out_size, void* d_ws, size_t ws_size,
                              hipStream_t stream) {
  const float* x      = (const float*)d_in[0];
  const float* t      = (const float*)d_in[1];
  const int*   ei     = (const int*)  d_in[2];
  const float* ew     = (const float*)d_in[3];
  const float* gn0_w  = (const float*)d_in[4];
  const float* gn0_b  = (const float*)d_in[5];
  const float* W1     = (const float*)d_in[6];
  const float* b1     = (const float*)d_in[7];
  const float* gn1_w  = (const float*)d_in[8];
  const float* gn1_b  = (const float*)d_in[9];
  const float* W2     = (const float*)d_in[10];
  const float* b2     = (const float*)d_in[11];
  const float* Wres   = (const float*)d_in[12];
  const float* te_w   = (const float*)d_in[13];
  const float* te_b   = (const float*)d_in[14];
  const float* gn2_w  = (const float*)d_in[15];
  const float* gn2_b  = (const float*)d_in[16];
  const float* Wa     = (const float*)d_in[17];
  const float* a_src  = (const float*)d_in[18];
  const float* a_dst  = (const float*)d_in[19];
  const float* a_edge = (const float*)d_in[20];
  const float* ba     = (const float*)d_in[21];

  const int N = in_sizes[0] / CIN;
  const int E = in_sizes[2] / 2;
  const int* row = ei;
  const int* col = ei + E;

  // ---- workspace layout (256B aligned) ----
  char* base = (char*)d_ws;
  size_t off = 0;
  auto alloc = [&](size_t bytes) -> char* {
    char* p = base + off;
    off = (off + bytes + 255) & ~(size_t)255;
    return p;
  };
  float*     B0    = (float*)alloc((size_t)N * COUT * 4);  // xw1 -> xw2 -> hA
  float*     B1    = (float*)alloc((size_t)N * COUT * 4);  // xres
  float*     B2    = (float*)alloc((size_t)N * COUT * 4);  // h1 -> h
  _Float16*  H0    = (_Float16*)alloc((size_t)N * COUT * 2); // h1n16 -> a16
  _Float16*  XN16  = (_Float16*)alloc((size_t)N * CIN * 2);
  _Float16*  X16   = (_Float16*)alloc((size_t)N * CIN * 2);
  _Float16*  W1p   = (_Float16*)alloc(CIN * COUT * 2);     // packed fragments
  _Float16*  Wrp   = (_Float16*)alloc(CIN * COUT * 2);
  _Float16*  W2p   = (_Float16*)alloc(COUT * COUT * 2);
  _Float16*  Wap   = (_Float16*)alloc(COUT * COUT * 2);
  float*     dinv  = (float*)alloc((size_t)N * 4);
  float*     tt    = (float*)alloc(COUT * 4);
  float*     als   = (float*)alloc((size_t)N * NHEAD * 4);
  float*     ald   = (float*)alloc((size_t)N * NHEAD * 4);
  unsigned*  mx    = (unsigned*)alloc((size_t)N * NHEAD * 4);
  float*     sden  = (float*)alloc((size_t)N * NHEAD * 4);
  unsigned*  fillb = (unsigned*)alloc(4);
  (void)ws_size; (void)n_in; (void)out_size;

  float* out = (float*)d_out;

  const int TPB = 256;
  auto cdiv = [](long long a, long long b) { return (int)((a + b - 1) / b); };

  // ---- weight packing + time embedding + degree norm ----
  k_pack_b<<<cdiv(CIN * COUT, TPB), TPB, 0, stream>>>(W1,   W1p, CIN,  COUT);
  k_pack_b<<<cdiv(CIN * COUT, TPB), TPB, 0, stream>>>(Wres, Wrp, CIN,  COUT);
  k_pack_b<<<cdiv(COUT * COUT, TPB), TPB, 0, stream>>>(W2,  W2p, COUT, COUT);
  k_pack_b<<<cdiv(COUT * COUT, TPB), TPB, 0, stream>>>(Wa,  Wap, COUT, COUT);
  k_time_embed<<<1, COUT, 0, stream>>>(t, te_w, te_b, tt);
  k_deg_init<<<cdiv(N, TPB), TPB, 0, stream>>>(dinv, fillb, N);
  k_deg_edge<<<cdiv(E, TPB), TPB, 0, stream>>>(col, ew, dinv, fillb, E);
  k_dinv<<<cdiv(N, TPB), TPB, 0, stream>>>(dinv, N);

  dim3 gemmGrid(cdiv(N, 64), COUT / 64);

  // ---- ResBlock layer 1: h1 = GCN(leaky(GN0(x)), W1) + b1 + tt ----
  k_gn64<<<cdiv((long long)N * CIN, TPB), TPB, 0, stream>>>(x, gn0_w, gn0_b, XN16, X16, N);
  k_gemm_wmma<CIN><<<gemmGrid, 128, 0, stream>>>(XN16, W1p, B0, N, COUT);   // xw1
  k_gemm_wmma<CIN><<<gemmGrid, 128, 0, stream>>>(X16,  Wrp, B1, N, COUT);   // xres
  k_gcn_init<<<cdiv((long long)N * COUT, TPB), TPB, 0, stream>>>(B2, B0, dinv, b1, tt, nullptr, N);
  k_gcn_edge<<<cdiv((long long)E * 32, TPB), TPB, 0, stream>>>(row, col, ew, dinv, B0, B2, E);

  // ---- ResBlock layer 2: h = GCN(leaky(GN1(h1)), W2) + b2 + x@Wres ----
  k_gn128<<<cdiv((long long)N * COUT, TPB), TPB, 0, stream>>>(B2, gn1_w, gn1_b, H0, N);
  k_gemm_wmma<COUT><<<gemmGrid, 128, 0, stream>>>(H0, W2p, B0, N, COUT);    // xw2
  k_gcn_init<<<cdiv((long long)N * COUT, TPB), TPB, 0, stream>>>(B2, B0, dinv, b2, nullptr, B1, N);
  k_gcn_edge<<<cdiv((long long)E * 32, TPB), TPB, 0, stream>>>(row, col, ew, dinv, B0, B2, E);

  // ---- AttnBlock: out = GAT(leaky(GN2(h))) ----
  k_gn128<<<cdiv((long long)N * COUT, TPB), TPB, 0, stream>>>(B2, gn2_w, gn2_b, H0, N);
  k_gemm_wmma<COUT><<<gemmGrid, 128, 0, stream>>>(H0, Wap, B0, N, COUT);    // hA
  k_attn_coef<<<cdiv((long long)N * NHEAD, TPB), TPB, 0, stream>>>(B0, a_src, a_dst, als, ald, N);
  k_attn_minit<<<cdiv((long long)N * NHEAD, TPB), TPB, 0, stream>>>(als, ald, a_edge, fillb, mx, N);
  k_attn_medge<<<cdiv((long long)E * NHEAD, TPB), TPB, 0, stream>>>(row, col, ew, als, ald, a_edge, mx, E);
  k_attn_sinit<<<cdiv((long long)N * NHEAD, TPB), TPB, 0, stream>>>(als, ald, a_edge, fillb, mx, sden, N);
  k_attn_sedge<<<cdiv((long long)E * NHEAD, TPB), TPB, 0, stream>>>(row, col, ew, als, ald, a_edge, mx, sden, E);
  k_attn_outinit<<<cdiv((long long)N * COUT, TPB), TPB, 0, stream>>>(B0, als, ald, a_edge, fillb, mx, sden, ba, out, N);
  k_attn_edge<<<cdiv((long long)E * 32, TPB), TPB, 0, stream>>>(row, col, ew, als, ald, a_edge, mx, sden, B0, out, E);
}
